// LeafLinearForest_87531433492896
// MI455X (gfx1250) — matrix operand order repaired and checked by reading, not compiled, not run on hardware
//
#include <hip/hip_runtime.h>

typedef __attribute__((ext_vector_type(16))) _Float16 v16h;
typedef __attribute__((ext_vector_type(8)))  _Float16 v8h;
typedef __attribute__((ext_vector_type(8)))  float    v8f;

#define NT      32          // trees
#define NDEPTH  4
#define NL      16          // leaves
#define NK      4
#define ND      256         // feature dim
#define NTHR    6           // thresholds per feature
#define NF0     1536        // D * NTH
#define NFIN    3072        // 2 * NF0
#define NB      16384       // batch
#define TDN     (NT*NDEPTH)     // 128 rows of sel
#define LEAFROWS (NT*NL*NK)     // 2048 rows of leaf_sel

// ---------------- block reductions (blockDim == 256) ----------------
__device__ __forceinline__ float blockSum256(float v, float* red) {
  const int tid = threadIdx.x;
  __syncthreads();
  red[tid] = v;
  __syncthreads();
  #pragma unroll
  for (int s = 128; s > 0; s >>= 1) {
    if (tid < s) red[tid] += red[tid + s];
    __syncthreads();
  }
  return red[0];
}
__device__ __forceinline__ float blockMax256(float v, float* red) {
  const int tid = threadIdx.x;
  __syncthreads();
  red[tid] = v;
  __syncthreads();
  #pragma unroll
  for (int s = 128; s > 0; s >>= 1) {
    if (tid < s) red[tid] = fmaxf(red[tid], red[tid + s]);
    __syncthreads();
  }
  return red[0];
}

// ---------------- sparsemax of sel_logits (128 rows x 3072) ----------------
// Writes selW[td,f] = sparsemax(z)[f] - sparsemax(z)[f+1536]  (f16, 128x1536)
// and selBias[td] = sum_f sparsemax(z)[f+1536]                (f32, 128)
__global__ __launch_bounds__(256)
void prep_sel_kernel(const float* __restrict__ logits,
                     _Float16* __restrict__ selW,
                     float* __restrict__ selBias) {
  __shared__ float z[NFIN];
  __shared__ float red[256];
  const int tid = threadIdx.x;
  const int row = blockIdx.x;
  const float* src = logits + (size_t)row * NFIN;
  for (int i = tid; i < NFIN; i += 256) z[i] = src[i];
  __syncthreads();

  float mx = -3.4e38f;
  for (int i = tid; i < NFIN; i += 256) mx = fmaxf(mx, z[i]);
  mx = blockMax256(mx, red);

  // sparsemax tau solves sum(relu(z - tau)) = 1; tau in [mx-1, mx]
  float lo = mx - 1.0f, hi = mx;
  for (int it = 0; it < 40; ++it) {
    const float tau = 0.5f * (lo + hi);
    float s = 0.0f;
    for (int i = tid; i < NFIN; i += 256) s += fmaxf(z[i] - tau, 0.0f);
    s = blockSum256(s, red);
    if (s > 1.0f) lo = tau; else hi = tau;
  }
  const float tau = 0.5f * (lo + hi);

  float bias = 0.0f;
  for (int i = tid; i < NF0; i += 256) {
    const float a = fmaxf(z[i]        - tau, 0.0f);
    const float c = fmaxf(z[i + NF0]  - tau, 0.0f);
    selW[(size_t)row * NF0 + i] = (_Float16)(a - c);
    bias += c;
  }
  bias = blockSum256(bias, red);
  if (tid == 0) selBias[row] = bias;
}

// ---------------- sparsemax of leaf_sel_logits (2048 rows x 256) ----------------
__global__ __launch_bounds__(256)
void prep_leafsel_kernel(const float* __restrict__ logits,
                         _Float16* __restrict__ leafW) {
  __shared__ float red[256];
  const int tid = threadIdx.x;
  const int row = blockIdx.x;
  const float zv = logits[(size_t)row * ND + tid];

  const float mx = blockMax256(zv, red);
  float lo = mx - 1.0f, hi = mx;
  for (int it = 0; it < 40; ++it) {
    const float tau = 0.5f * (lo + hi);
    const float s = blockSum256(fmaxf(zv - tau, 0.0f), red);
    if (s > 1.0f) lo = tau; else hi = tau;
  }
  const float tau = 0.5f * (lo + hi);
  leafW[(size_t)row * ND + tid] = (_Float16)fmaxf(zv - tau, 0.0f);
}

// ---------------- fused main kernel ----------------
// One block = 16 batch rows, 8 waves. Wave w owns trees 4w..4w+3.
// GEMM1: p[16 x 16cols] over K=1536 (features built in LDS, never hit HBM)
// GEMM2: xleaf tiles [16 x 16] over K=256, 16 tiles per wave
__global__ __launch_bounds__(256)
void forest_main_kernel(const float* __restrict__ x,
                        const float* __restrict__ thresholds,
                        const float* __restrict__ steepness,
                        const _Float16* __restrict__ selW,
                        const float* __restrict__ selBias,
                        const _Float16* __restrict__ leafW,
                        const float* __restrict__ leaf_w,
                        const float* __restrict__ leaf_b,
                        float* __restrict__ out) {
  __shared__ _Float16 fT[16 * NF0];   // 48 KB feature tile (f16)
  __shared__ _Float16 xT[16 * ND];    // 8 KB x tile (f16)
  __shared__ float lds_p[8][16][16];  // per-wave routing probs
  __shared__ float ldsOut[8][16];     // per-wave partial outputs

  const int tid  = threadIdx.x;
  const int m0   = blockIdx.x * 16;
  const int wave = tid >> 5;
  const int lane = tid & 31;
  const int h    = lane >> 4;   // K-half selector (WMMA lane layout)
  const int nn   = lane & 15;   // row (A) / col (B,C) selector

  // Build x tile (f16)
  for (int i = tid; i < 16 * ND; i += 256) {
    const int m = i >> 8, d = i & 255;
    xT[i] = (_Float16)x[(size_t)(m0 + m) * ND + d];
  }
  // Build feature tile: f[m, d*6+j] = sigmoid(steep*(x - thr))
  for (int i = tid; i < 16 * NF0; i += 256) {
    const int m = i / NF0;
    const int f = i - m * NF0;
    const int d = f / NTHR;
    const int j = f - d * NTHR;
    const float xv = x[(size_t)(m0 + m) * ND + d];
    const float t  = steepness[d * NTHR + j] * (xv - thresholds[d * NTHR + j]);
    fT[i] = (_Float16)(1.0f / (1.0f + __expf(-t)));
  }
  __syncthreads();

  // ---- GEMM1: routing logits p (this wave's 16 td-columns) ----
  v8f cp = {};
  {
    const _Float16* Abase = &fT[nn * NF0 + 16 * h];
    const _Float16* Bbase = selW + (size_t)(16 * wave + nn) * NF0 + 16 * h;
    for (int k0 = 0; k0 < NF0; k0 += 32) {
      __builtin_prefetch(Bbase + k0 + 256, 0, 1);
      const v8h a0 = *(const v8h*)(Abase + k0);
      const v8h a1 = *(const v8h*)(Abase + k0 + 8);
      const v16h a = __builtin_shufflevector(a0, a1,
          0,1,2,3,4,5,6,7,8,9,10,11,12,13,14,15);
      const v16h b = *(const v16h*)(Bbase + k0);
      cp = __builtin_amdgcn_wmma_f32_16x16x32_f16(false, a, false, b,
                                                  (short)0, cp, false, false);
    }
  }
  // p = clip(p + selBias, 1e-6, 1-1e-6) -> LDS
  {
    const float sb = selBias[16 * wave + nn];
    #pragma unroll
    for (int r = 0; r < 8; ++r) {
      float pv = cp[r] + sb;
      pv = fminf(fmaxf(pv, 1e-6f), 1.0f - 1e-6f);
      lds_p[wave][r + 8 * h][nn] = pv;
    }
  }
  __syncthreads();

  // Preload GEMM2 A operand (x rows, K=256 -> 8 chunks) once
  v16h a2[8];
  {
    const _Float16* Abase = &xT[nn * ND + 16 * h];
    #pragma unroll
    for (int kk = 0; kk < 8; ++kk) {
      const v8h a0 = *(const v8h*)(Abase + kk * 32);
      const v8h a1 = *(const v8h*)(Abase + kk * 32 + 8);
      a2[kk] = __builtin_shufflevector(a0, a1,
          0,1,2,3,4,5,6,7,8,9,10,11,12,13,14,15);
    }
  }

  v8f acc = {};
  for (int tl = 0; tl < 4; ++tl) {
    const int t = wave * 4 + tl;
    #pragma unroll
    for (int q = 0; q < 4; ++q) {
      // xleaf tile: columns = (leaf l = 4q + nn>>2, k = nn&3)
      v8f cx = {};
      const _Float16* Bbase = leafW + (size_t)(t * 64 + q * 16 + nn) * ND + 16 * h;
      #pragma unroll
      for (int kk = 0; kk < 8; ++kk) {
        const v16h b = *(const v16h*)(Bbase + kk * 32);
        cx = __builtin_amdgcn_wmma_f32_16x16x32_f16(false, a2[kk], false, b,
                                                    (short)0, cx, false, false);
      }
      const int   l   = q * 4 + (nn >> 2);
      const float wv  = leaf_w[t * 64 + q * 16 + nn];       // leaf_w[t,l,k]
      const float lb4 = 4.0f * leaf_b[t * 16 + l];          // K * leaf_b[t,l]
      #pragma unroll
      for (int r = 0; r < 8; ++r) {
        // sum over k (4 adjacent lanes), then add 4*leaf_b
        float y = cx[r] * wv;
        y += __shfl_xor(y, 1, 32);
        y += __shfl_xor(y, 2, 32);
        y += lb4;
        // routing probability of leaf l: bit d selects p_d vs (1-p_d)
        float pl = 1.0f;
        #pragma unroll
        for (int d = 0; d < 4; ++d) {
          const float pd = lds_p[wave][r + 8 * h][4 * tl + d];
          pl *= ((l >> d) & 1) ? pd : (1.0f - pd);
        }
        acc[r] += pl * y;   // each leaf counted 4x (k-group lanes) -> /4 later
      }
    }
  }

  // Reduce across 16 columns (lanes 0..15 and 16..31 independently)
  #pragma unroll
  for (int r = 0; r < 8; ++r) {
    float v = acc[r];
    v += __shfl_xor(v, 1, 32);
    v += __shfl_xor(v, 2, 32);
    v += __shfl_xor(v, 4, 32);
    v += __shfl_xor(v, 8, 32);
    acc[r] = v;
  }
  if (nn == 0) {
    #pragma unroll
    for (int r = 0; r < 8; ++r) ldsOut[wave][r + 8 * h] = acc[r] * 0.25f;
  }
  __syncthreads();
  if (tid < 16) {
    float s = 0.0f;
    #pragma unroll
    for (int w = 0; w < 8; ++w) s += ldsOut[w][tid];
    out[m0 + tid] = s;
  }
}

// ---------------- launcher ----------------
extern "C" void kernel_launch(void* const* d_in, const int* in_sizes, int n_in,
                              void* d_out, int out_size, void* d_ws, size_t ws_size,
                              hipStream_t stream) {
  const float* x               = (const float*)d_in[0];
  const float* thresholds      = (const float*)d_in[1];
  const float* steepness       = (const float*)d_in[2];
  const float* sel_logits      = (const float*)d_in[3];
  const float* leaf_sel_logits = (const float*)d_in[4];
  const float* leaf_w          = (const float*)d_in[5];
  const float* leaf_b          = (const float*)d_in[6];
  float* out = (float*)d_out;

  // workspace layout (all offsets 512B-aligned)
  char* ws = (char*)d_ws;
  _Float16* selW    = (_Float16*)(ws);                        // 128*1536*2 = 393216 B
  float*    selBias = (float*)(ws + 393216);                  // 512 B
  _Float16* leafW   = (_Float16*)(ws + 393216 + 512);         // 2048*256*2 = 1 MB

  prep_sel_kernel<<<TDN, 256, 0, stream>>>(sel_logits, selW, selBias);
  prep_leafsel_kernel<<<LEAFROWS, 256, 0, stream>>>(leaf_sel_logits, leafW);
  forest_main_kernel<<<NB / 16, 256, 0, stream>>>(
      x, thresholds, steepness, selW, selBias, leafW, leaf_w, leaf_b, out);
}
